// WARP_62672162783299
// MI455X (gfx1250) — compile-verified, matrix operand order checked by reading, not run on hardware
//
#include <hip/hip_runtime.h>
#include <hip/hip_bf16.h>
#include <utility>

typedef __attribute__((ext_vector_type(16))) _Float16 v16h;
typedef __attribute__((ext_vector_type(8)))  _Float16 v8h;
typedef __attribute__((ext_vector_type(8)))  float    v8f;
typedef __attribute__((ext_vector_type(4)))  unsigned int u32x4;
typedef __attribute__((ext_vector_type(8)))  int      i32x8;
typedef __attribute__((ext_vector_type(4)))  int      i32x4;

#define PPIX   9216     // 96*96 pixels
#define TFRM   12
#define DTHETA 11015
#define DFEAT  1024
#define INNER_LR 0.01f
#define INNER_STEPS 64
#define NSPLIT 32       // split-K factor for weight-grad GEMMs
#define USE_TDM 1       // stage weight tile into LDS via Tensor Data Mover

// ---------------------------------------------------------------------------
// TDM: DMA one contiguous 64x64 f16 weight tile (8192 B) into LDS offset 0.
// D# per CDNA5 ISA ch.8: group0 {count=1, lds_addr, global_addr[56:0], type=2},
// group1 {data_size=1(2B), tile 4096 x 1, tensor_dim0=4096, stride=4096}.
// ---------------------------------------------------------------------------
__device__ __forceinline__ void tdm_load_weights(const _Float16* gptr)
{
#if USE_TDM
    unsigned long long ga = (unsigned long long)(uintptr_t)gptr;
    u32x4 g0;
    g0[0] = 1u;                                        // count=1 valid descriptor
    g0[1] = 0u;                                        // lds_addr = 0 (only LDS object)
    g0[2] = (unsigned)(ga & 0xFFFFFFFFu);              // global_addr[31:0]
    g0[3] = (unsigned)((ga >> 32) & 0x01FFFFFFu)       // global_addr[56:32]
          | (2u << 30);                                // type = 2 ("image")
    i32x8 g1;
    g1[0] = 0x00010000;          // workgroup_mask=0, data_size=1 (2 bytes)
    g1[1] = (int)(4096u << 16);  // tensor_dim0 = 4096 (bits 79:48, low half)
    g1[2] = (int)(1u << 16);     // tensor_dim0 hi = 0, tensor_dim1 = 1
    g1[3] = (int)(4096u << 16);  // tensor_dim1 hi = 0, tile_dim0 = 4096
    g1[4] = 1;                   // tile_dim1 = 1, tile_dim2 = 0
    g1[5] = 4096;                // tensor_dim0_stride = 4096 (elements)
    g1[6] = 0;                   // stride hi / tensor_dim1_stride
    g1[7] = 0;
    i32x4 z4 = {0, 0, 0, 0};
#if defined(__clang_major__) && (__clang_major__ >= 23)
    i32x8 z8 = {0, 0, 0, 0, 0, 0, 0, 0};
    __builtin_amdgcn_tensor_load_to_lds(g0, g1, z4, z4, z8, 0);
#else
    __builtin_amdgcn_tensor_load_to_lds(g0, g1, z4, z4, 0);
#endif
    __builtin_amdgcn_s_wait_tensorcnt(0);
#endif
}

// ---------------------------------------------------------------------------
// Fourier encoding: encH [P x 64] row-major f16 (cols 34..63 zero)
//                   encT [64 x P] transposed copy for weight-grad GEMM
// ---------------------------------------------------------------------------
__global__ void k_encode(const float* __restrict__ coords,
                         _Float16* __restrict__ encH, _Float16* __restrict__ encT)
{
    int p = blockIdx.x * blockDim.x + threadIdx.x;
    if (p >= PPIX) return;
    float x = coords[p * 2 + 0];
    float y = coords[p * 2 + 1];
    const float PI = 3.14159265358979323846f;
    float vals[64];
    vals[0] = x; vals[1] = y;
#pragma unroll
    for (int f = 0; f < 8; ++f) {
        float fr = (float)(1 << f) * PI;
        vals[2 + f]  = sinf(x * fr);
        vals[10 + f] = sinf(y * fr);
        vals[18 + f] = cosf(x * fr);
        vals[26 + f] = cosf(y * fr);
    }
#pragma unroll
    for (int j = 34; j < 64; ++j) vals[j] = 0.0f;
    _Float16* e = encH + (size_t)p * 64;
#pragma unroll
    for (int j = 0; j < 64; ++j) {
        _Float16 h = (_Float16)vals[j];
        e[j] = h;
        encT[(size_t)j * PPIX + p] = h;
    }
}

// ---------------------------------------------------------------------------
// WMMA GEMM with TDM-staged weights:
//   A  : [M x K] row-major f16 (global, contiguous b128 fragment loads)
//   Bc : [N x K] row-major f16 (== B^T), 64x64 region staged into LDS by TDM,
//        fragments read as two ds_load_b128 per tile.
// C[M x N] = act(A @ B + bias).  NT = N/16 tiles per wave (A-fragment reuse).
// 4 waves/block -> 64 rows. Requires ldb == 64, K <= 64 (true for all layers).
// ---------------------------------------------------------------------------
template <int NT>
__global__ void k_gemm_nt(const _Float16* __restrict__ A, int lda,
                          const _Float16* __restrict__ Bc,
                          const float* __restrict__ bias,
                          const float* __restrict__ mask, int ldm,
                          int relu,
                          float* __restrict__ C32, int ldc,
                          _Float16* __restrict__ C16, int ldc16,
                          _Float16* __restrict__ C16T, int ldct,
                          int K)
{
    __shared__ _Float16 ldsB[64 * 64];   // only LDS object -> offset 0

    const int lane  = threadIdx.x & 31;
    const int wave  = threadIdx.x >> 5;
    const int m0    = blockIdx.x * 64 + wave * 16;
    const int sub   = lane & 15;
    const int kbase = (lane < 16) ? 0 : 8;

#if USE_TDM
    if (threadIdx.x < 32) tdm_load_weights(Bc);
#else
    for (int i = threadIdx.x; i < 4096 / 8; i += blockDim.x)
        ((v8h*)ldsB)[i] = ((const v8h*)Bc)[i];
#endif
    __syncthreads();

    v8f acc[NT];
#pragma unroll
    for (int j = 0; j < NT; ++j)
        acc[j] = (v8f){0.f, 0.f, 0.f, 0.f, 0.f, 0.f, 0.f, 0.f};

    for (int kk = 0; kk < K; kk += 32) {
        const _Float16* ap = A + (size_t)(m0 + sub) * lda + (kk + kbase);
        v8h a0 = *(const v8h*)(ap);
        v8h a1 = *(const v8h*)(ap + 16);
        v16h af = __builtin_shufflevector(a0, a1, 0, 1, 2, 3, 4, 5, 6, 7,
                                          8, 9, 10, 11, 12, 13, 14, 15);
        __builtin_prefetch(ap + 32, 0, 0);
#pragma unroll
        for (int j = 0; j < NT; ++j) {
            const _Float16* bp = ldsB + (j * 16 + sub) * 64 + (kk + kbase);
            v8h b0 = *(const v8h*)(bp);
            v8h b1 = *(const v8h*)(bp + 16);
            v16h bf = __builtin_shufflevector(b0, b1, 0, 1, 2, 3, 4, 5, 6, 7,
                                              8, 9, 10, 11, 12, 13, 14, 15);
            acc[j] = __builtin_amdgcn_wmma_f32_16x16x32_f16(false, af, false, bf,
                                                            (short)0, acc[j], false, false);
        }
    }

    const int mb = m0 + ((lane < 16) ? 0 : 8);
#pragma unroll
    for (int j = 0; j < NT; ++j) {
        const int n = j * 16 + sub;
        const float bv = bias ? bias[n] : 0.0f;
#pragma unroll
        for (int r = 0; r < 8; ++r) {
            const int m = mb + r;
            float v = acc[j][r] + bv;
            if (relu) v = fmaxf(v, 0.0f);
            if (mask) v = (mask[(size_t)m * ldm + n] > 0.0f) ? v : 0.0f;
            if (C32)  C32[(size_t)m * ldc + n] = v;
            if (C16)  C16[(size_t)m * ldc16 + n] = (_Float16)v;
            if (C16T) C16T[(size_t)n * ldct + m] = (_Float16)v;
        }
    }
}

// ---------------------------------------------------------------------------
// Weight-gradient GEMM: gW[M x 64] = AT[M x K] @ Xin, K = PPIX, split-K.
//   AT : dX^T  [M x K] row-major (M = doutPad 16 or 64)
//   Bc : Xin^T [64 x K] row-major (streamed once; stays global b128)
// grid.x = NSPLIT slices; blockDim = (M/16)*32. Deterministic partial slices.
// ---------------------------------------------------------------------------
__global__ void k_gemm_atb(const _Float16* __restrict__ AT, int lda,
                           const _Float16* __restrict__ Bc, int ldb,
                           float* __restrict__ Cpart,   // [NSPLIT][64*64]
                           int kPerSplit, int K)
{
    const int lane  = threadIdx.x & 31;
    const int wave  = threadIdx.x >> 5;
    const int m0    = wave * 16;
    const int sub   = lane & 15;
    const int kbase = (lane < 16) ? 0 : 8;
    int k0 = blockIdx.x * kPerSplit;
    int k1 = k0 + kPerSplit; if (k1 > K) k1 = K;

    v8f acc[4];
#pragma unroll
    for (int j = 0; j < 4; ++j)
        acc[j] = (v8f){0.f, 0.f, 0.f, 0.f, 0.f, 0.f, 0.f, 0.f};

    for (int kk = k0; kk < k1; kk += 32) {
        const _Float16* ap = AT + (size_t)(m0 + sub) * lda + (kk + kbase);
        v8h a0 = *(const v8h*)(ap);
        v8h a1 = *(const v8h*)(ap + 16);
        v16h af = __builtin_shufflevector(a0, a1, 0, 1, 2, 3, 4, 5, 6, 7,
                                          8, 9, 10, 11, 12, 13, 14, 15);
        __builtin_prefetch(ap + 32, 0, 0);
#pragma unroll
        for (int j = 0; j < 4; ++j) {
            const _Float16* bp = Bc + (size_t)(j * 16 + sub) * ldb + (kk + kbase);
            v8h b0 = *(const v8h*)(bp);
            v8h b1 = *(const v8h*)(bp + 16);
            v16h bf = __builtin_shufflevector(b0, b1, 0, 1, 2, 3, 4, 5, 6, 7,
                                              8, 9, 10, 11, 12, 13, 14, 15);
            acc[j] = __builtin_amdgcn_wmma_f32_16x16x32_f16(false, af, false, bf,
                                                            (short)0, acc[j], false, false);
        }
    }

    float* slice = Cpart + (size_t)blockIdx.x * 4096;
    const int mb = m0 + ((lane < 16) ? 0 : 8);
#pragma unroll
    for (int j = 0; j < 4; ++j)
#pragma unroll
        for (int r = 0; r < 8; ++r)
            slice[(size_t)(mb + r) * 64 + (j * 16 + sub)] = acc[j][r];
}

__global__ void k_reduce_splits(const float* __restrict__ part, float* __restrict__ outp, int n)
{
    int i = blockIdx.x * blockDim.x + threadIdx.x;
    if (i >= n) return;
    float s = 0.f;
#pragma unroll 4
    for (int sN = 0; sN < NSPLIT; ++sN) s += part[(size_t)sN * 4096 + i];
    outp[i] = s;
}

// Bias gradient: row sum of transposed activation grad (coalesced) + LDS reduce
__global__ void k_colsum_t(const _Float16* __restrict__ AT, int Kd, float* __restrict__ outp)
{
    __shared__ float red[256];
    float s = 0.f;
    const _Float16* row = AT + (size_t)blockIdx.x * Kd;
    for (int k = threadIdx.x; k < Kd; k += 256) s += (float)row[k];
    red[threadIdx.x] = s;
    __syncthreads();
    for (int off = 128; off > 0; off >>= 1) {
        if (threadIdx.x < off) red[threadIdx.x] += red[threadIdx.x + off];
        __syncthreads();
    }
    if (threadIdx.x == 0) outp[blockIdx.x] = red[0];
}

__global__ void k_zero(float* __restrict__ p, int n)
{
    int i = blockIdx.x * blockDim.x + threadIdx.x;
    if (i < n) p[i] = 0.f;
}

// ---------------------------------------------------------------------------
// Pack flat theta -> padded layouts. W16[r*64+c]=W[r][c] (Bc for forward);
// WT16[c*64+r]=W[r][c] (Bc for backward dX).
// ---------------------------------------------------------------------------
__global__ void k_pack(const float* __restrict__ theta, int off, int din, int dout,
                       float* __restrict__ W32, float* __restrict__ b32,
                       _Float16* __restrict__ W16, _Float16* __restrict__ WT16)
{
    int idx = blockIdx.x * blockDim.x + threadIdx.x;
    if (idx < 4096) {
        int r = idx >> 6, c = idx & 63;
        float v = (r < dout && c < din) ? theta[off + r * din + c] : 0.0f;
        if (W32) W32[idx] = v;
        W16[idx] = (_Float16)v;
        WT16[c * 64 + r] = (_Float16)v;
    } else if (idx < 4096 + 64) {
        int i = idx - 4096;
        b32[i] = (i < dout) ? theta[off + din * dout + i] : 0.0f;
    }
}

__global__ void k_unpack(float* __restrict__ outTheta, int off, int din, int dout,
                         const float* __restrict__ W32, const float* __restrict__ b32)
{
    int idx = blockIdx.x * blockDim.x + threadIdx.x;
    int nw = din * dout;
    if (idx < nw) {
        int r = idx / din, c = idx % din;
        outTheta[off + idx] = W32[r * 64 + c];
    } else if (idx < nw + dout) {
        outTheta[off + idx] = b32[idx - nw];
    }
}

__global__ void k_update(float* __restrict__ W32, float* __restrict__ b32,
                         const float* __restrict__ gW, const float* __restrict__ gb,
                         _Float16* __restrict__ W16, _Float16* __restrict__ WT16)
{
    int idx = blockIdx.x * blockDim.x + threadIdx.x;
    if (idx < 4096) {
        float w = W32[idx] - INNER_LR * gW[idx];
        W32[idx] = w;
        int r = idx >> 6, c = idx & 63;
        W16[idx] = (_Float16)w;
        WT16[c * 64 + r] = (_Float16)w;
    } else if (idx < 4096 + 64) {
        int i = idx - 4096;
        b32[i] -= INNER_LR * gb[i];
    }
}

// ---------------------------------------------------------------------------
// L1 loss gradient through sigmoid render head.
// out32 [P x 16]; dOut [P x 32] f16 row-major; dOutT [16 x P] f16 transposed.
// ---------------------------------------------------------------------------
__device__ __forceinline__ float sigm(float x) { return 1.0f / (1.0f + expf(-x)); }

__global__ void k_loss_grad(const float* __restrict__ out32, const float* __restrict__ gt0,
                            _Float16* __restrict__ dOut, _Float16* __restrict__ dOutT)
{
    int p = blockIdx.x * blockDim.x + threadIdx.x;
    if (p >= PPIX) return;
    const float* o = out32 + (size_t)p * 16;
    const float gscale = 1.0f / (float)(PPIX * 3);
    float alpha = sigm(o[6]);
    float d[7]; float da = 0.f;
#pragma unroll
    for (int c = 0; c < 3; ++c) {
        float fg = sigm(o[c]);
        float bg = sigm(o[3 + c]);
        float pred = alpha * fg + (1.f - alpha) * bg;
        float r = pred - gt0[p * 3 + c];
        float g = (r > 0.f) ? gscale : ((r < 0.f) ? -gscale : 0.f);
        d[c]     = g * alpha * fg * (1.f - fg);
        d[3 + c] = g * (1.f - alpha) * bg * (1.f - bg);
        da      += g * (fg - bg);
    }
    d[6] = da * alpha * (1.f - alpha);
    _Float16* dp = dOut + (size_t)p * 32;
#pragma unroll
    for (int j = 0; j < 32; ++j) dp[j] = (_Float16)((j < 7) ? d[j] : 0.0f);
#pragma unroll
    for (int j = 0; j < 16; ++j)
        dOutT[(size_t)j * PPIX + p] = (_Float16)((j < 7) ? d[j] : 0.0f);
}

// Render + select frame_t = use_gt ? gt : pred
__global__ void k_render(const float* __restrict__ out32, const float* __restrict__ gtFrame,
                         const int* __restrict__ use_gt, int t,
                         float* __restrict__ predOut, float* __restrict__ frameOut)
{
    int p = blockIdx.x * blockDim.x + threadIdx.x;
    if (p >= PPIX) return;
    const float* o = out32 + (size_t)p * 16;
    float alpha = sigm(o[6]);
    bool ug = (use_gt[t] != 0);
#pragma unroll
    for (int c = 0; c < 3; ++c) {
        float pred = alpha * sigm(o[c]) + (1.f - alpha) * sigm(o[3 + c]);
        predOut[(size_t)p * 3 + c] = pred;
        frameOut[(size_t)p * 3 + c] = ug ? gtFrame[(size_t)p * 3 + c] : pred;
    }
}

// 3x3 stride-2 pad-1 conv + relu. hwc=1: input layout HWC; else CHW.
__global__ void k_conv(const float* __restrict__ in, int Ci, int Hi, int Wi, int hwc,
                       const float* __restrict__ wgt, const float* __restrict__ b,
                       float* __restrict__ outp, int Co, int Ho, int Wo)
{
    int idx = blockIdx.x * blockDim.x + threadIdx.x;
    int total = Co * Ho * Wo;
    if (idx >= total) return;
    int ox = idx % Wo;
    int oy = (idx / Wo) % Ho;
    int co = idx / (Wo * Ho);
    float s = b[co];
    for (int ci = 0; ci < Ci; ++ci) {
#pragma unroll
        for (int ky = 0; ky < 3; ++ky) {
            int iy = oy * 2 - 1 + ky;
            if (iy < 0 || iy >= Hi) continue;
#pragma unroll
            for (int kx = 0; kx < 3; ++kx) {
                int ix = ox * 2 - 1 + kx;
                if (ix < 0 || ix >= Wi) continue;
                float iv = hwc ? in[((size_t)iy * Wi + ix) * Ci + ci]
                               : in[((size_t)ci * Hi + iy) * Wi + ix];
                s += iv * wgt[(((size_t)co * Ci + ci) * 3 + ky) * 3 + kx];
            }
        }
    }
    outp[idx] = fmaxf(s, 0.0f);
}

// Wave-per-row f32 matvec (HBM-bound: phi_lw 406 MB, psi_lw 38 MB)
__global__ void k_matvec(const float* __restrict__ Mx, const float* __restrict__ x,
                         const float* __restrict__ bias, float* __restrict__ y,
                         int R, int Kd)
{
    int lane = threadIdx.x & 31;
    int row  = blockIdx.x * (blockDim.x >> 5) + (threadIdx.x >> 5);
    if (row >= R) return;
    const float* mr = Mx + (size_t)row * Kd;
    float s = 0.f;
    for (int k = lane; k < Kd; k += 32) s += mr[k] * x[k];
#pragma unroll
    for (int off = 16; off > 0; off >>= 1) s += __shfl_xor(s, off, 32);
    if (lane == 0) y[row] = s + (bias ? bias[row] : 0.0f);
}

// theta_next = A @ theta + B @ dx (12 x 485 MB of A traffic: pure HBM bound)
__global__ void k_theta_step(const float* __restrict__ Am, const float* __restrict__ Bm,
                             const float* __restrict__ th, const float* __restrict__ dx,
                             float* __restrict__ thNext)
{
    int lane = threadIdx.x & 31;
    int row  = blockIdx.x * (blockDim.x >> 5) + (threadIdx.x >> 5);
    if (row >= DTHETA) return;
    float s = 0.f;
    const float* ar = Am + (size_t)row * DTHETA;
    for (int k = lane; k < DTHETA; k += 32) s += ar[k] * th[k];
    const float* br = Bm + (size_t)row * DFEAT;
    for (int k = lane; k < DFEAT; k += 32) s += br[k] * dx[k];
#pragma unroll
    for (int off = 16; off > 0; off >>= 1) s += __shfl_xor(s, off, 32);
    if (lane == 0) thNext[row] = s;
}

__global__ void k_dx(const float* __restrict__ ft, const float* __restrict__ fp,
                     float* __restrict__ dx)
{
    int i = blockIdx.x * blockDim.x + threadIdx.x;
    if (i < DFEAT) dx[i] = (ft[i] - fp[i]) * (1.0f / 32.0f); // 1/sqrt(1024)
}

// ===========================================================================
extern "C" void kernel_launch(void* const* d_in, const int* in_sizes, int n_in,
                              void* d_out, int out_size, void* d_ws, size_t ws_size,
                              hipStream_t stream)
{
    (void)in_sizes; (void)n_in; (void)out_size; (void)ws_size;

    const float* ref     = (const float*)d_in[0];
    const float* coords  = (const float*)d_in[1];
    const float* theta0f = (const float*)d_in[2];
    const float* Amat    = (const float*)d_in[3];
    const float* Bmat    = (const float*)d_in[4];
    const float* phi_cw[3] = {(const float*)d_in[5], (const float*)d_in[7], (const float*)d_in[9]};
    const float* phi_cb[3] = {(const float*)d_in[6], (const float*)d_in[8], (const float*)d_in[10]};
    const float* phi_lw = (const float*)d_in[11];
    const float* phi_lb = (const float*)d_in[12];
    const float* psi_cw[3] = {(const float*)d_in[13], (const float*)d_in[15], (const float*)d_in[17]};
    const float* psi_cb[3] = {(const float*)d_in[14], (const float*)d_in[16], (const float*)d_in[18]};
    const float* psi_lw = (const float*)d_in[19];
    const float* psi_lb = (const float*)d_in[20];
    const int*   use_gt = (const int*)d_in[21];

    static const int LDIN[4]  = {34, 64, 64, 64};
    static const int LDOUT[4] = {64, 64, 64, 7};
    static const int LOFF[4]  = {0, 2240, 6400, 10560};

    // ---- workspace carve-up (~26 MB) ----
    char* wsp = (char*)d_ws;
    size_t off = 0;
    auto walloc = [&](size_t bytes) -> char* {
        char* p = wsp + off;
        off = (off + bytes + 255) & ~(size_t)255;
        return p;
    };
    _Float16* encH   = (_Float16*)walloc((size_t)PPIX * 64 * 2);
    _Float16* encT   = (_Float16*)walloc((size_t)PPIX * 64 * 2);
    _Float16* h16[3];  for (int i = 0; i < 3; ++i) h16[i]  = (_Float16*)walloc((size_t)PPIX * 64 * 2);
    _Float16* h16T[3]; for (int i = 0; i < 3; ++i) h16T[i] = (_Float16*)walloc((size_t)PPIX * 64 * 2);
    float*    h32[3];  for (int i = 0; i < 3; ++i) h32[i]  = (float*)walloc((size_t)PPIX * 64 * 4);
    float*    out32  = (float*)walloc((size_t)PPIX * 16 * 4);
    _Float16* dOut16 = (_Float16*)walloc((size_t)PPIX * 32 * 2);
    _Float16* dOutT  = (_Float16*)walloc((size_t)PPIX * 16 * 2);
    _Float16* dX16[3]; for (int i = 0; i < 3; ++i) dX16[i] = (_Float16*)walloc((size_t)PPIX * 64 * 2);
    _Float16* dXT[3];  for (int i = 0; i < 3; ++i) dXT[i]  = (_Float16*)walloc((size_t)PPIX * 64 * 2);
    float*    W32[4];  for (int i = 0; i < 4; ++i) W32[i]  = (float*)walloc(4096 * 4);
    float*    b32[4];  for (int i = 0; i < 4; ++i) b32[i]  = (float*)walloc(64 * 4);
    _Float16* W16[4];  for (int i = 0; i < 4; ++i) W16[i]  = (_Float16*)walloc(4096 * 2);
    _Float16* WT16[4]; for (int i = 0; i < 4; ++i) WT16[i] = (_Float16*)walloc(4096 * 2);
    float* gradsBase = (float*)walloc((4 * 4096 + 4 * 64) * 4);
    float* gW[4]; float* gb[4];
    for (int i = 0; i < 4; ++i) { gW[i] = gradsBase + i * 4096; gb[i] = gradsBase + 4 * 4096 + i * 64; }
    float* partial  = (float*)walloc((size_t)NSPLIT * 4096 * 4);
    float* thetaA   = (float*)walloc(11264 * 4);
    float* thetaBuf = (float*)walloc(11264 * 4);
    float* featA    = (float*)walloc(DFEAT * 4);
    float* featB    = (float*)walloc(DFEAT * 4);
    float* dxb      = (float*)walloc(DFEAT * 4);
    float* frameBuf = (float*)walloc((size_t)PPIX * 3 * 4);
    float* cb1      = (float*)walloc((size_t)16 * 48 * 48 * 4);
    float* cb2      = (float*)walloc((size_t)32 * 24 * 24 * 4);
    float* cb3      = (float*)walloc((size_t)64 * 12 * 12 * 4);

    float* predOut  = (float*)d_out;
    float* thetaOpt = (float*)d_out + (size_t)TFRM * PPIX * 3;

    const dim3 b256(256);
    const dim3 gP((PPIX + 255) / 256);
    const dim3 gemmB(128);
    const dim3 gM(PPIX / 64);
    const int  kps = PPIX / NSPLIT;   // 288

    // ================= Phase A: inner GD (theta_opt) =================
    k_encode<<<gP, b256, 0, stream>>>(coords, encH, encT);
    for (int l = 0; l < 4; ++l)
        k_pack<<<dim3(17), b256, 0, stream>>>(theta0f, LOFF[l], LDIN[l], LDOUT[l],
                                              W32[l], b32[l], W16[l], WT16[l]);

    for (int it = 0; it < INNER_STEPS; ++it) {
        // ---- forward (Bc = W16 row-major == column-major W^T) ----
        k_gemm_nt<4><<<gM, gemmB, 0, stream>>>(encH, 64, W16[0], b32[0],
            (const float*)nullptr, 0, 1, h32[0], 64, h16[0], 64, h16T[0], PPIX, 64);
        k_gemm_nt<4><<<gM, gemmB, 0, stream>>>(h16[0], 64, W16[1], b32[1],
            (const float*)nullptr, 0, 1, h32[1], 64, h16[1], 64, h16T[1], PPIX, 64);
        k_gemm_nt<4><<<gM, gemmB, 0, stream>>>(h16[1], 64, W16[2], b32[2],
            (const float*)nullptr, 0, 1, h32[2], 64, h16[2], 64, h16T[2], PPIX, 64);
        k_gemm_nt<1><<<gM, gemmB, 0, stream>>>(h16[2], 64, W16[3], b32[3],
            (const float*)nullptr, 0, 0, out32, 16, (_Float16*)nullptr, 0,
            (_Float16*)nullptr, 0, 64);

        k_loss_grad<<<gP, b256, 0, stream>>>(out32, ref, dOut16, dOutT);
        k_zero<<<dim3((4 * 4096 + 4 * 64 + 255) / 256), b256, 0, stream>>>(gradsBase, 4 * 4096 + 4 * 64);

        // ---- backward (Bc = WT16 for dX) ----
        // dX3 = (dOut @ W4) * (h3>0)
        k_gemm_nt<4><<<gM, gemmB, 0, stream>>>(dOut16, 32, WT16[3],
            (const float*)nullptr, h32[2], 64, 0, (float*)nullptr, 0,
            dX16[2], 64, dXT[2], PPIX, 32);
        k_gemm_atb<<<dim3(NSPLIT), dim3(32), 0, stream>>>(dOutT, PPIX, h16T[2], PPIX,
            partial, kps, PPIX);
        k_reduce_splits<<<dim3((16 * 64 + 255) / 256), b256, 0, stream>>>(partial, gW[3], 16 * 64);
        k_colsum_t<<<dim3(16), b256, 0, stream>>>(dOutT, PPIX, gb[3]);

        // dX2 = (dX3 @ W3) * (h2>0)
        k_gemm_nt<4><<<gM, gemmB, 0, stream>>>(dX16[2], 64, WT16[2],
            (const float*)nullptr, h32[1], 64, 0, (float*)nullptr, 0,
            dX16[1], 64, dXT[1], PPIX, 64);
        k_gemm_atb<<<dim3(NSPLIT), dim3(128), 0, stream>>>(dXT[2], PPIX, h16T[1], PPIX,
            partial, kps, PPIX);
        k_reduce_splits<<<dim3((4096 + 255) / 256), b256, 0, stream>>>(partial, gW[2], 4096);
        k_colsum_t<<<dim3(64), b256, 0, stream>>>(dXT[2], PPIX, gb[2]);

        // dX1 = (dX2 @ W2) * (h1>0)
        k_gemm_nt<4><<<gM, gemmB, 0, stream>>>(dX16[1], 64, WT16[1],
            (const float*)nullptr, h32[0], 64, 0, (float*)nullptr, 0,
            dX16[0], 64, dXT[0], PPIX, 64);
        k_gemm_atb<<<dim3(NSPLIT), dim3(128), 0, stream>>>(dXT[1], PPIX, h16T[0], PPIX,
            partial, kps, PPIX);
        k_reduce_splits<<<dim3((4096 + 255) / 256), b256, 0, stream>>>(partial, gW[1], 4096);
        k_colsum_t<<<dim3(64), b256, 0, stream>>>(dXT[1], PPIX, gb[1]);

        // gW0 = dX1^T @ enc
        k_gemm_atb<<<dim3(NSPLIT), dim3(128), 0, stream>>>(dXT[0], PPIX, encT, PPIX,
            partial, kps, PPIX);
        k_reduce_splits<<<dim3((4096 + 255) / 256), b256, 0, stream>>>(partial, gW[0], 4096);
        k_colsum_t<<<dim3(64), b256, 0, stream>>>(dXT[0], PPIX, gb[0]);

        for (int l = 0; l < 4; ++l)
            k_update<<<dim3(17), b256, 0, stream>>>(W32[l], b32[l], gW[l], gb[l], W16[l], WT16[l]);
    }
    for (int l = 0; l < 4; ++l) {
        int n = LDIN[l] * LDOUT[l] + LDOUT[l];
        k_unpack<<<dim3((n + 255) / 256), b256, 0, stream>>>(thetaOpt, LOFF[l], LDIN[l], LDOUT[l],
                                                             W32[l], b32[l]);
    }

    // ===== Phase B: theta_0 = phi-CNN(frame0); featPrev = psi-CNN(frame0) =====
    k_conv<<<dim3((16 * 48 * 48 + 255) / 256), b256, 0, stream>>>(ref, 3, 96, 96, 1,
        phi_cw[0], phi_cb[0], cb1, 16, 48, 48);
    k_conv<<<dim3((32 * 24 * 24 + 255) / 256), b256, 0, stream>>>(cb1, 16, 48, 48, 0,
        phi_cw[1], phi_cb[1], cb2, 32, 24, 24);
    k_conv<<<dim3((64 * 12 * 12 + 255) / 256), b256, 0, stream>>>(cb2, 32, 24, 24, 0,
        phi_cw[2], phi_cb[2], cb3, 64, 12, 12);
    k_matvec<<<dim3((DTHETA + 3) / 4), dim3(128), 0, stream>>>(phi_lw, cb3, phi_lb, thetaA, DTHETA, 9216);

    k_conv<<<dim3((16 * 48 * 48 + 255) / 256), b256, 0, stream>>>(ref, 3, 96, 96, 1,
        psi_cw[0], psi_cb[0], cb1, 16, 48, 48);
    k_conv<<<dim3((32 * 24 * 24 + 255) / 256), b256, 0, stream>>>(cb1, 16, 48, 48, 0,
        psi_cw[1], psi_cb[1], cb2, 32, 24, 24);
    k_conv<<<dim3((64 * 12 * 12 + 255) / 256), b256, 0, stream>>>(cb2, 32, 24, 24, 0,
        psi_cw[2], psi_cb[2], cb3, 64, 12, 12);
    k_matvec<<<dim3((DFEAT + 3) / 4), dim3(128), 0, stream>>>(psi_lw, cb3, psi_lb, featA, DFEAT, 9216);

    // ================= Phase C: scan over T frames =================
    float* cur = thetaA;  float* nxt = thetaBuf;
    float* fprev = featA; float* fcur = featB;
    for (int t = 0; t < TFRM; ++t) {
        for (int l = 0; l < 4; ++l)
            k_pack<<<dim3(17), b256, 0, stream>>>(cur, LOFF[l], LDIN[l], LDOUT[l],
                                                  (float*)nullptr, b32[l], W16[l], WT16[l]);
        k_gemm_nt<4><<<gM, gemmB, 0, stream>>>(encH, 64, W16[0], b32[0],
            (const float*)nullptr, 0, 1, (float*)nullptr, 0, h16[0], 64,
            (_Float16*)nullptr, 0, 64);
        k_gemm_nt<4><<<gM, gemmB, 0, stream>>>(h16[0], 64, W16[1], b32[1],
            (const float*)nullptr, 0, 1, (float*)nullptr, 0, h16[1], 64,
            (_Float16*)nullptr, 0, 64);
        k_gemm_nt<4><<<gM, gemmB, 0, stream>>>(h16[1], 64, W16[2], b32[2],
            (const float*)nullptr, 0, 1, (float*)nullptr, 0, h16[2], 64,
            (_Float16*)nullptr, 0, 64);
        k_gemm_nt<1><<<gM, gemmB, 0, stream>>>(h16[2], 64, W16[3], b32[3],
            (const float*)nullptr, 0, 0, out32, 16, (_Float16*)nullptr, 0,
            (_Float16*)nullptr, 0, 64);

        int gti = (t + 1 < TFRM) ? (t + 1) : (TFRM - 1);
        const float* gtf = ref + (size_t)gti * PPIX * 3;
        k_render<<<gP, b256, 0, stream>>>(out32, gtf, use_gt, t,
                                          predOut + (size_t)t * PPIX * 3, frameBuf);

        k_conv<<<dim3((16 * 48 * 48 + 255) / 256), b256, 0, stream>>>(frameBuf, 3, 96, 96, 1,
            psi_cw[0], psi_cb[0], cb1, 16, 48, 48);
        k_conv<<<dim3((32 * 24 * 24 + 255) / 256), b256, 0, stream>>>(cb1, 16, 48, 48, 0,
            psi_cw[1], psi_cb[1], cb2, 32, 24, 24);
        k_conv<<<dim3((64 * 12 * 12 + 255) / 256), b256, 0, stream>>>(cb2, 32, 24, 24, 0,
            psi_cw[2], psi_cb[2], cb3, 64, 12, 12);
        k_matvec<<<dim3((DFEAT + 3) / 4), dim3(128), 0, stream>>>(psi_lw, cb3, psi_lb, fcur, DFEAT, 9216);

        k_dx<<<dim3((DFEAT + 255) / 256), b256, 0, stream>>>(fcur, fprev, dxb);
        k_theta_step<<<dim3((DTHETA + 3) / 4), dim3(128), 0, stream>>>(Amat, Bmat, cur, dxb, nxt);

        std::swap(cur, nxt);
        std::swap(fprev, fcur);
    }
}
